// Self_Attention_523986010551
// MI455X (gfx1250) — matrix operand order repaired and checked by reading, not compiled, not run on hardware
//
#include <hip/hip_runtime.h>
#include <hip/hip_bf16.h>

// ---------------------------------------------------------------------------
// Self-attention, B=8, N=2048, D=1024, f32 in/out, bf16 WMMA compute.
// softmax over axis=1 (query axis) per the reference.
// ---------------------------------------------------------------------------

typedef __attribute__((ext_vector_type(16))) __bf16 v16bf;
typedef __attribute__((ext_vector_type(8)))  __bf16 v8bf;
typedef __attribute__((ext_vector_type(4)))  __bf16 v4bf;
typedef __attribute__((ext_vector_type(8)))  float  v8f;

#define BATCH 8
#define SEQ   2048
#define DIM   1024

// ---------------------------------------------------------------------------
// f32 -> bf16 elementwise cast, 4-wide
// ---------------------------------------------------------------------------
__global__ void cast_f32_to_bf16(const float* __restrict__ src,
                                 __bf16* __restrict__ dst, int n4) {
  int i = blockIdx.x * blockDim.x + threadIdx.x;
  if (i < n4) {
    float4 v = ((const float4*)src)[i];
    v4bf o;
    o[0] = (__bf16)v.x; o[1] = (__bf16)v.y;
    o[2] = (__bf16)v.z; o[3] = (__bf16)v.w;
    ((v4bf*)dst)[i] = o;
  }
}

// ---------------------------------------------------------------------------
// W[k][n] (f32, DxD) -> Wt[n][k] (bf16) via LDS tile
// ---------------------------------------------------------------------------
__global__ void transpose_cast_bf16(const float* __restrict__ W,
                                    __bf16* __restrict__ Wt, int D) {
  __shared__ float tile[32][33];
  int n = blockIdx.x * 32 + threadIdx.x;
  int k = blockIdx.y * 32 + threadIdx.y;
  tile[threadIdx.y][threadIdx.x] = W[(size_t)k * D + n];
  __syncthreads();
  int on = blockIdx.x * 32 + threadIdx.y;   // output row = n
  int ok = blockIdx.y * 32 + threadIdx.x;   // output col = k
  Wt[(size_t)on * D + ok] = (__bf16)tile[threadIdx.x][threadIdx.y];
}

// ---------------------------------------------------------------------------
// WMMA fragment loads per CDNA5 ISA 16-bit layouts (wave32).
// A 16x32 (MxK): lanes 0-15 row M=lane, halves K[0..7]/K[16..23];
//                lanes 16-31 same row, halves K[8..15]/K[24..31].
// ---------------------------------------------------------------------------
__device__ __forceinline__ v16bf load_frag_a(const __bf16* __restrict__ base,
                                             int ldk, int lane) {
  const int r   = lane & 15;
  const int sel = (lane >> 4) & 1;
  const __bf16* p = base + (size_t)r * ldk + sel * 8;
  v8bf lo = *(const v8bf*)p;
  v8bf hi = *(const v8bf*)(p + 16);
  v16bf f;
#pragma unroll
  for (int i = 0; i < 8; ++i) { f[i] = lo[i]; f[i + 8] = hi[i]; }
  return f;
}

// B 32x16 (KxN), fed from Bt[N][K] row-major:
// lanes 0-15 column N=lane holds K[0..7]/K[16..23]; lanes 16-31 hold K[8..15]/K[24..31].
__device__ __forceinline__ v16bf load_frag_b(const __bf16* __restrict__ base,
                                             int ldk, int lane) {
  const int c   = lane & 15;
  const int sel = (lane >> 4) & 1;
  const __bf16* p = base + (size_t)c * ldk + sel * 8;
  v8bf lo = *(const v8bf*)p;
  v8bf hi = *(const v8bf*)(p + 16);
  v16bf f;
#pragma unroll
  for (int i = 0; i < 8; ++i) { f[i] = lo[i]; f[i + 8] = hi[i]; }
  return f;
}

__device__ __forceinline__ v8f wmma_bf16(v16bf a, v16bf b, v8f c) {
  // (neg_a, A, neg_b, B, c_mod, C, reuse_a, reuse_b)
  return __builtin_amdgcn_wmma_f32_16x16x32_bf16(false, a, false, b,
                                                 (short)0, c, false, false);
}

// ---------------------------------------------------------------------------
// C[M,N] = A[M,K] * Bt[N,K]^T   (both bf16 row-major, f32 accumulate)
// block = 256 threads = 8 waves (4 in M x 2 in N); block tile 128x64,
// wave tile 32x32 (2x2 WMMA accumulators), K-step 32.
// MODE 0: f32 row-major store (batched via cBS)
// MODE 1: bf16 row-major store
// MODE 2: bf16 transposed store with batch fold: row t -> vT[b][d][t&2047]
// ---------------------------------------------------------------------------
template <int MODE>
__global__ void gemm_wmma_tn(const __bf16* __restrict__ A,
                             const __bf16* __restrict__ Bt,
                             void* __restrict__ Cv,
                             int K,
                             unsigned long long aBS,
                             unsigned long long bBS,
                             unsigned long long cBS,
                             int ldc) {
  const int lane = threadIdx.x & 31;
  const int wave = threadIdx.x >> 5;
  const int wm = wave >> 1;
  const int wn = wave & 1;
  const int m0 = blockIdx.y * 128 + wm * 32;
  const int n0 = blockIdx.x * 64 + wn * 32;

  A  += (size_t)blockIdx.z * aBS;
  Bt += (size_t)blockIdx.z * bBS;

  v8f acc[2][2] = {};

  for (int k0 = 0; k0 < K; k0 += 32) {
    v16bf a0 = load_frag_a(A + (size_t)m0 * K + k0, K, lane);
    v16bf a1 = load_frag_a(A + (size_t)(m0 + 16) * K + k0, K, lane);
    v16bf b0 = load_frag_b(Bt + (size_t)n0 * K + k0, K, lane);
    v16bf b1 = load_frag_b(Bt + (size_t)(n0 + 16) * K + k0, K, lane);
    acc[0][0] = wmma_bf16(a0, b0, acc[0][0]);
    acc[0][1] = wmma_bf16(a0, b1, acc[0][1]);
    acc[1][0] = wmma_bf16(a1, b0, acc[1][0]);
    acc[1][1] = wmma_bf16(a1, b1, acc[1][1]);
  }

  // C/D layout: lanes 0-15 -> M = j,   N = lane
  //             lanes 16-31 -> M = j+8, N = lane-16
  const int col = lane & 15;
  const int rb  = (lane < 16) ? 0 : 8;

  if constexpr (MODE == 0) {
    float* C = (float*)Cv + (size_t)blockIdx.z * cBS;
#pragma unroll
    for (int i = 0; i < 2; ++i)
#pragma unroll
      for (int jn = 0; jn < 2; ++jn)
#pragma unroll
        for (int j = 0; j < 8; ++j)
          C[(size_t)(m0 + i * 16 + rb + j) * ldc + (n0 + jn * 16 + col)] =
              acc[i][jn][j];
  } else if constexpr (MODE == 1) {
    __bf16* C = (__bf16*)Cv;
#pragma unroll
    for (int i = 0; i < 2; ++i)
#pragma unroll
      for (int jn = 0; jn < 2; ++jn)
#pragma unroll
        for (int j = 0; j < 8; ++j)
          C[(size_t)(m0 + i * 16 + rb + j) * ldc + (n0 + jn * 16 + col)] =
              (__bf16)acc[i][jn][j];
  } else {
    // v projection -> vT : token t = flat row, d = col.
    // vT[b][d][t&2047], per-batch stride D*SEQ
    __bf16* C = (__bf16*)Cv;
#pragma unroll
    for (int i = 0; i < 2; ++i)
#pragma unroll
      for (int jn = 0; jn < 2; ++jn)
#pragma unroll
        for (int j = 0; j < 8; ++j) {
          int t = m0 + i * 16 + rb + j;
          int d = n0 + jn * 16 + col;
          int b = t >> 11;
          int nn = t & (SEQ - 1);
          C[(size_t)b * ((size_t)DIM * SEQ) + (size_t)d * SEQ + nn] =
              (__bf16)acc[i][jn][j];
        }
  }
}

// ---------------------------------------------------------------------------
// Masked online softmax over the QUERY axis (axis=1): for each (b, m) column
// of scores[b][n][m], normalize over n. One thread per column; reads across
// threads are contiguous in m (coalesced). Writes attn as bf16.
// ---------------------------------------------------------------------------
__global__ void col_softmax_mask(const float* __restrict__ S,
                                 const int* __restrict__ Mk,
                                 __bf16* __restrict__ Attn, int N) {
  const int m = blockIdx.x * blockDim.x + threadIdx.x;
  const size_t base = (size_t)blockIdx.y * N * N;
  const float NEGV = -1.0e12f;

  float mx = -3.0e38f, sum = 0.0f;
  for (int n = 0; n < N; ++n) {
    size_t idx = base + (size_t)n * N + m;
    float v = (Mk[idx] == 0) ? NEGV : S[idx];
    float nm = fmaxf(mx, v);
    sum = sum * __expf(mx - nm) + __expf(v - nm);
    mx = nm;
  }
  float inv = 1.0f / sum;
  for (int n = 0; n < N; ++n) {
    size_t idx = base + (size_t)n * N + m;
    float v = (Mk[idx] == 0) ? NEGV : S[idx];
    Attn[idx] = (__bf16)(__expf(v - mx) * inv);
  }
}

// ---------------------------------------------------------------------------
// Launch
// ---------------------------------------------------------------------------
extern "C" void kernel_launch(void* const* d_in, const int* in_sizes, int n_in,
                              void* d_out, int out_size, void* d_ws,
                              size_t ws_size, hipStream_t stream) {
  (void)in_sizes; (void)n_in; (void)out_size; (void)ws_size;

  const float* x  = (const float*)d_in[0];   // [B, N, D]
  const float* Wq = (const float*)d_in[1];   // [D, D]
  const float* Wk = (const float*)d_in[2];
  const float* Wv = (const float*)d_in[3];
  const int*   mk = (const int*)d_in[4];     // [B, N, N]
  float* out = (float*)d_out;                // [B, N, D]

  char* ws = (char*)d_ws;
  const size_t TOK = (size_t)BATCH * SEQ;            // 16384 tokens
  __bf16* xb     = (__bf16*)(ws);                    // TOK*DIM bf16      (32 MiB)
  __bf16* wqt    = (__bf16*)(ws + 33554432ull);      // DIM*DIM bf16
  __bf16* wkt    = (__bf16*)(ws + 35651584ull);
  __bf16* wvt    = (__bf16*)(ws + 37748736ull);
  __bf16* qb     = (__bf16*)(ws + 39845888ull);      // TOK*DIM bf16
  __bf16* kb     = (__bf16*)(ws + 73400320ull);      // TOK*DIM bf16
  __bf16* vtb    = (__bf16*)(ws + 106954752ull);     // B * DIM * SEQ bf16 (vT)
  float*  scores = (float*)(ws + 140509184ull);      // B*N*N f32        (128 MiB)
  __bf16* attnb  = (__bf16*)(ws + 274726912ull);     // B*N*N bf16       (64 MiB)

  // 1) casts
  {
    int n4 = (int)(TOK * DIM / 4);
    cast_f32_to_bf16<<<(n4 + 255) / 256, 256, 0, stream>>>(x, xb, n4);
    dim3 tg(32, 32), tb(32, 32);
    transpose_cast_bf16<<<tg, tb, 0, stream>>>(Wq, wqt, DIM);
    transpose_cast_bf16<<<tg, tb, 0, stream>>>(Wk, wkt, DIM);
    transpose_cast_bf16<<<tg, tb, 0, stream>>>(Wv, wvt, DIM);
  }

  // 2) projections: q = x*Wq, k = x*Wk (bf16 row-major), v -> vT (bf16)
  {
    dim3 grid(DIM / 64, (unsigned)(TOK / 128), 1);
    gemm_wmma_tn<1><<<grid, 256, 0, stream>>>(xb, wqt, qb, DIM, 0ull, 0ull,
                                              0ull, DIM);
    gemm_wmma_tn<1><<<grid, 256, 0, stream>>>(xb, wkt, kb, DIM, 0ull, 0ull,
                                              0ull, DIM);
    gemm_wmma_tn<2><<<grid, 256, 0, stream>>>(xb, wvt, vtb, DIM, 0ull, 0ull,
                                              0ull, 0);
  }

  // 3) scores[b] = q[b] * k[b]^T   (f32)
  {
    dim3 grid(SEQ / 64, SEQ / 128, BATCH);
    gemm_wmma_tn<0><<<grid, 256, 0, stream>>>(
        qb, kb, scores, DIM,
        (unsigned long long)SEQ * DIM, (unsigned long long)SEQ * DIM,
        (unsigned long long)SEQ * SEQ, SEQ);
  }

  // 4) mask + softmax over query axis -> attn (bf16)
  {
    dim3 grid(SEQ / 256, BATCH);
    col_softmax_mask<<<grid, 256, 0, stream>>>(scores, mk, attnb, SEQ);
  }

  // 5) out[b] = attn[b] * v[b]   (Bt = vT[b], f32 store to d_out)
  {
    dim3 grid(DIM / 64, SEQ / 128, BATCH);
    gemm_wmma_tn<0><<<grid, 256, 0, stream>>>(
        attnb, vtb, out, SEQ,
        (unsigned long long)SEQ * SEQ, (unsigned long long)DIM * SEQ,
        (unsigned long long)SEQ * DIM, DIM);
  }
}
